// MultiHeadAttention_22935125360739
// MI455X (gfx1250) — matrix operand order repaired and checked by reading, not compiled, not run on hardware
//
#include <hip/hip_runtime.h>
#include <hip/hip_bf16.h>

#define B_  2
#define T_  4096
#define D_  768
#define H_  12
#define HD_ 64

typedef __attribute__((ext_vector_type(4)))  __bf16 bf16x4;
typedef __attribute__((ext_vector_type(8)))  __bf16 bf16x8;
typedef __attribute__((ext_vector_type(16))) __bf16 bf16x16;
typedef __attribute__((ext_vector_type(8)))  float  f32x8;

static __device__ inline f32x8 zero8() {
  f32x8 v = {0.f,0.f,0.f,0.f,0.f,0.f,0.f,0.f};
  return v;
}
static __device__ inline bf16x16 cat8(bf16x8 lo, bf16x8 hi) {
  return __builtin_shufflevector(lo, hi, 0,1,2,3,4,5,6,7,8,9,10,11,12,13,14,15);
}
static __device__ inline f32x8 wmma_bf16(bf16x16 a, bf16x16 b, f32x8 c) {
  // D = A(16x32 bf16) * B(32x16 bf16) + C(16x16 f32)
  return __builtin_amdgcn_wmma_f32_16x16x32_bf16(false, a, false, b, (short)0, c,
                                                 false, false);
}

// ---------------------------------------------------------------------------
// fp32 -> bf16 conversion (vectorized 4-wide)
// ---------------------------------------------------------------------------
__global__ void cvt_f32_to_bf16(const float* __restrict__ in,
                                __bf16* __restrict__ out, int n4) {
  int i = blockIdx.x * blockDim.x + threadIdx.x;
  if (i < n4) {
    float4 v = ((const float4*)in)[i];
    bf16x4 o = { (__bf16)v.x, (__bf16)v.y, (__bf16)v.z, (__bf16)v.w };
    *(bf16x4*)(out + 4 * i) = o;
  }
}

// ---------------------------------------------------------------------------
// QKV projection: [8192,768] x W^T -> per-head layouts.
//   z=0 -> Q [B,H,T,64], z=1 -> K [B,H,T,64], z=2 -> V transposed [B,H,64,T]
// One wave computes a 16x64 output tile (4 WMMA accumulators), K-loop step 32.
// ---------------------------------------------------------------------------
__global__ void __launch_bounds__(128) qkv_gemm(
    const __bf16* __restrict__ X,
    const __bf16* __restrict__ Wq, const __bf16* __restrict__ Wk,
    const __bf16* __restrict__ Wv,
    __bf16* __restrict__ Qo, __bf16* __restrict__ Ko, __bf16* __restrict__ Vt) {
  const int z = blockIdx.z;
  const __bf16* W = (z == 0) ? Wq : (z == 1 ? Wk : Wv);
  const int wave = threadIdx.x >> 5;
  const int lane = threadIdx.x & 31;
  const int ln = lane & 15, half = lane >> 4;
  const int row0 = (blockIdx.x * 4 + wave) * 16;
  const int h = blockIdx.y;        // this 64-col tile == one head
  const int col0 = h * 64;
  const int klo = half * 16;       // A-fragment k sub-chunk select

  f32x8 acc[4] = {zero8(), zero8(), zero8(), zero8()};
  const __bf16* xr = X + (size_t)(row0 + ln) * D_;
  for (int k0 = 0; k0 < D_; k0 += 32) {
    bf16x16 a = cat8(*(const bf16x8*)(xr + k0 + klo),
                     *(const bf16x8*)(xr + k0 + klo + 8));
#pragma unroll
    for (int nt = 0; nt < 4; nt++) {
      const __bf16* wr = W + (size_t)(col0 + nt * 16 + ln) * D_ + k0 + half * 16;
      acc[nt] = wmma_bf16(a, *(const bf16x16*)wr, acc[nt]);
    }
  }
#pragma unroll
  for (int nt = 0; nt < 4; nt++) {
#pragma unroll
    for (int r = 0; r < 8; r++) {
      int row = row0 + r + 8 * half;       // C layout: M = r + 8*(lane>=16)
      int b = row >> 12, t = row & (T_ - 1);
      int c = nt * 16 + ln;                // col within head
      __bf16 v = (__bf16)acc[nt][r];
      if (z == 0)      Qo[(((size_t)b * H_ + h) * T_ + t) * HD_ + c] = v;
      else if (z == 1) Ko[(((size_t)b * H_ + h) * T_ + t) * HD_ + c] = v;
      else             Vt[(((size_t)b * H_ + h) * HD_ + c) * T_ + t] = v;
    }
  }
}

// ---------------------------------------------------------------------------
// Flash attention: one wave per (b, h, 16-query tile); causal key chunks of 32.
// ---------------------------------------------------------------------------
__global__ void __launch_bounds__(128) attn_fwd(
    const __bf16* __restrict__ Q, const __bf16* __restrict__ K,
    const __bf16* __restrict__ Vt, __bf16* __restrict__ ctx) {
  __shared__ __attribute__((aligned(32))) __bf16 lds[4][16 * 32];
  const int wave = threadIdx.x >> 5, lane = threadIdx.x & 31;
  const int ln = lane & 15, half = lane >> 4;
  const int wid = blockIdx.x * 4 + wave;
  const int QT = T_ / 16;
  const int b = wid / (H_ * QT);
  const int rem = wid % (H_ * QT);
  const int h = rem / QT;
  const int q0 = (rem % QT) * 16;

  const __bf16* Qb = Q + (((size_t)b * H_ + h) * T_) * HD_;
  const __bf16* Kb = K + (((size_t)b * H_ + h) * T_) * HD_;
  const __bf16* Vb = Vt + (((size_t)b * H_ + h) * HD_) * T_;
  const int klo = half * 16;

  // Q A-fragments for head-dim chunks d=[0,32) and d=[32,64)
  const __bf16* qr = Qb + (size_t)(q0 + ln) * HD_;
  bf16x16 aq0 = cat8(*(const bf16x8*)(qr + klo),      *(const bf16x8*)(qr + klo + 8));
  bf16x16 aq1 = cat8(*(const bf16x8*)(qr + 32 + klo), *(const bf16x8*)(qr + 32 + klo + 8));

  float mr[8], lr[8];
#pragma unroll
  for (int r = 0; r < 8; r++) { mr[r] = -__builtin_inff(); lr[r] = 0.f; }
  f32x8 o[4] = {zero8(), zero8(), zero8(), zero8()};
  __bf16* pl = lds[wave];

  const int nChunk = (q0 + 47) >> 5;     // ceil((q0+16)/32), <= T/32
  for (int kc = 0; kc < nChunk; kc++) {
    const int kb = kc * 32;
    // --- scores: two 16x16 tiles over 32 keys, K-dim = 64 head dims ---
    f32x8 s0 = zero8(), s1 = zero8();
    {
      const __bf16* kr = Kb + (size_t)(kb + ln) * HD_ + half * 16;
      s0 = wmma_bf16(aq0, *(const bf16x16*)kr, s0);
      s0 = wmma_bf16(aq1, *(const bf16x16*)(kr + 32), s0);
    }
    {
      const __bf16* kr = Kb + (size_t)(kb + 16 + ln) * HD_ + half * 16;
      s1 = wmma_bf16(aq0, *(const bf16x16*)kr, s1);
      s1 = wmma_bf16(aq1, *(const bf16x16*)(kr + 32), s1);
    }
    if (kc + 1 < nChunk)
      __builtin_prefetch(Kb + (size_t)(kb + 32 + ln) * HD_, 0, 0);

    // --- causal mask + scale + online softmax (row = r + 8*half) ---
    float alpha[8];
#pragma unroll
    for (int r = 0; r < 8; r++) {
      const int rowg = q0 + r + 8 * half;
      const int key0 = kb + ln, key1 = kb + 16 + ln;
      float v0 = (key0 <= rowg) ? s0[r] * 0.125f : -__builtin_inff();
      float v1 = (key1 <= rowg) ? s1[r] * 0.125f : -__builtin_inff();
      float cm = fmaxf(v0, v1);
      cm = fmaxf(cm, __shfl_xor(cm, 1, 32));
      cm = fmaxf(cm, __shfl_xor(cm, 2, 32));
      cm = fmaxf(cm, __shfl_xor(cm, 4, 32));
      cm = fmaxf(cm, __shfl_xor(cm, 8, 32));
      float nm = fmaxf(mr[r], cm);           // finite: key kb always unmasked
      float al = __expf(mr[r] - nm);
      float p0 = __expf(v0 - nm);
      float p1 = __expf(v1 - nm);
      float rs = p0 + p1;
      rs += __shfl_xor(rs, 1, 32);
      rs += __shfl_xor(rs, 2, 32);
      rs += __shfl_xor(rs, 4, 32);
      rs += __shfl_xor(rs, 8, 32);
      lr[r] = lr[r] * al + rs;
      mr[r] = nm;
      alpha[r] = al;
      const int m = r + 8 * half;            // stage P (C-layout -> LDS)
      pl[m * 32 + ln]      = (__bf16)p0;
      pl[m * 32 + 16 + ln] = (__bf16)p1;
    }
    asm volatile("s_wait_dscnt 0" ::: "memory");   // intra-wave LDS transpose
    const __bf16* pr = pl + ln * 32;               // reload as A-fragment
    bf16x16 pa = cat8(*(const bf16x8*)(pr + klo), *(const bf16x8*)(pr + klo + 8));
    asm volatile("" ::: "memory");

    // --- O = O*alpha + P(16x32) x V(32x64) ---
#pragma unroll
    for (int nt = 0; nt < 4; nt++) {
#pragma unroll
      for (int r = 0; r < 8; r++) o[nt][r] *= alpha[r];
      const __bf16* vr = Vb + (size_t)(nt * 16 + ln) * T_ + kb + half * 16;
      o[nt] = wmma_bf16(pa, *(const bf16x16*)vr, o[nt]);
    }
  }

  // --- normalize and store ctx [B,T,D] bf16 ---
#pragma unroll
  for (int nt = 0; nt < 4; nt++)
#pragma unroll
    for (int r = 0; r < 8; r++) {
      int m = r + 8 * half;
      float v = o[nt][r] / lr[r];
      ctx[((size_t)(b * T_ + q0 + m)) * D_ + h * HD_ + nt * 16 + ln] = (__bf16)v;
    }
}

// ---------------------------------------------------------------------------
// Output projection: out = ctx @ Wo^T + bo (fp32 out)
// ---------------------------------------------------------------------------
__global__ void __launch_bounds__(128) out_gemm(
    const __bf16* __restrict__ X, const __bf16* __restrict__ Wo,
    const float* __restrict__ bo, float* __restrict__ out) {
  const int wave = threadIdx.x >> 5;
  const int lane = threadIdx.x & 31;
  const int ln = lane & 15, half = lane >> 4;
  const int row0 = (blockIdx.x * 4 + wave) * 16;
  const int col0 = blockIdx.y * 64;
  const int klo = half * 16;

  f32x8 acc[4] = {zero8(), zero8(), zero8(), zero8()};
  const __bf16* xr = X + (size_t)(row0 + ln) * D_;
  for (int k0 = 0; k0 < D_; k0 += 32) {
    bf16x16 a = cat8(*(const bf16x8*)(xr + k0 + klo),
                     *(const bf16x8*)(xr + k0 + klo + 8));
#pragma unroll
    for (int nt = 0; nt < 4; nt++) {
      const __bf16* wr = Wo + (size_t)(col0 + nt * 16 + ln) * D_ + k0 + half * 16;
      acc[nt] = wmma_bf16(a, *(const bf16x16*)wr, acc[nt]);
    }
  }
#pragma unroll
  for (int nt = 0; nt < 4; nt++) {
#pragma unroll
    for (int r = 0; r < 8; r++) {
      int row = row0 + r + 8 * half;
      int col = col0 + nt * 16 + ln;
      out[(size_t)row * D_ + col] = acc[nt][r] + bo[col];
    }
  }
}

// ---------------------------------------------------------------------------
extern "C" void kernel_launch(void* const* d_in, const int* in_sizes, int n_in,
                              void* d_out, int out_size, void* d_ws, size_t ws_size,
                              hipStream_t stream) {
  (void)in_sizes; (void)n_in; (void)out_size; (void)ws_size;
  const float* x  = (const float*)d_in[0];
  const float* Wq = (const float*)d_in[1];
  const float* Wk = (const float*)d_in[2];
  const float* Wv = (const float*)d_in[3];
  const float* Wo = (const float*)d_in[4];
  const float* bo = (const float*)d_in[5];
  float* out = (float*)d_out;

  const size_t NX = (size_t)B_ * T_ * D_;       // 6291456
  const size_t NW = (size_t)D_ * D_;            // 589824
  const size_t NH = (size_t)B_ * H_ * T_ * HD_; // 6291456

  __bf16* ws  = (__bf16*)d_ws;                  // ~67.6 MB bf16 scratch
  __bf16* xb  = ws;
  __bf16* wqb = xb + NX;
  __bf16* wkb = wqb + NW;
  __bf16* wvb = wkb + NW;
  __bf16* wob = wvb + NW;
  __bf16* Qw  = wob + NW;
  __bf16* Kw  = Qw + NH;
  __bf16* Vtw = Kw + NH;
  __bf16* ctx = Vtw + NH;

  const int thr = 256;
  cvt_f32_to_bf16<<<(int)((NX / 4 + thr - 1) / thr), thr, 0, stream>>>(x,  xb,  (int)(NX / 4));
  cvt_f32_to_bf16<<<(int)((NW / 4 + thr - 1) / thr), thr, 0, stream>>>(Wq, wqb, (int)(NW / 4));
  cvt_f32_to_bf16<<<(int)((NW / 4 + thr - 1) / thr), thr, 0, stream>>>(Wk, wkb, (int)(NW / 4));
  cvt_f32_to_bf16<<<(int)((NW / 4 + thr - 1) / thr), thr, 0, stream>>>(Wv, wvb, (int)(NW / 4));
  cvt_f32_to_bf16<<<(int)((NW / 4 + thr - 1) / thr), thr, 0, stream>>>(Wo, wob, (int)(NW / 4));

  // 8192 rows / (4 waves * 16) = 128 blocks; 768/64 = 12 head tiles; z = Q/K/V
  qkv_gemm<<<dim3(128, 12, 3), 128, 0, stream>>>(xb, wqb, wkb, wvb, Qw, Kw, Vtw);

  // B*H*(T/16) = 6144 waves / 4 per block = 1536 blocks
  attn_fwd<<<dim3(1536), 128, 0, stream>>>(Qw, Kw, Vtw, ctx);

  out_gemm<<<dim3(128, 12), 128, 0, stream>>>(ctx, wob, bo, out);
}
